// SWINTransformerBlock_76836964925524
// MI455X (gfx1250) — matrix operand order repaired and checked by reading, not compiled, not run on hardware
//
#include <hip/hip_runtime.h>
#include <hip/hip_bf16.h>
#include <math.h>

// ---------------- types ----------------
typedef __attribute__((ext_vector_type(16))) __bf16       bf16x16;
typedef __attribute__((ext_vector_type(8)))  float        f32x8;
typedef __attribute__((ext_vector_type(4)))  unsigned int u32x4;

union Frag { bf16x16 v; u32x4 u[2]; };

// ---------------- problem constants ----------------
// B=16, H=W=64, C=256, WIN=8, SHIFT=4, NH=8, HD=32, N=64, HIDDEN=1024
#define TOK   65536   // B*H*W tokens
#define CH    256
#define QKVN  768
#define HIDN  1024
#define NWIN  1024    // B * 64 windows

#define VT_STRIDE 72  // padded LDS strides (144B = 9*16B: b128-aligned,
#define PS_STRIDE 72  // and l16*36 mod 64 hits 16 distinct banks)

// ---------------- helpers ----------------
__device__ __forceinline__ unsigned short f2bf(float f) {
  unsigned u = __float_as_uint(f);
  u += 0x7fffu + ((u >> 16) & 1u);            // round-to-nearest-even
  return (unsigned short)(u >> 16);
}

// window-token index r -> pixel index after +SHIFT roll (used both for the
// LN1 gather (h[y0]=x[y0+4]) and the output scatter (final[y0+4]=pre[y0])).
__device__ __forceinline__ unsigned swin_pix(int r) {
  int win = r >> 6, t = r & 63;
  int b = win >> 6, wi = win & 63;
  int wh = wi >> 3, ww = wi & 7;
  int i = t >> 3,  j = t & 7;
  int y = (wh * 8 + i + 4) & 63;
  int x = (ww * 8 + j + 4) & 63;
  return ((unsigned)(b * 64 + y) * 64 + x);
}

// ---------------- f32 -> bf16 convert ----------------
__global__ void cvt_kernel(const float* __restrict__ s,
                           unsigned short* __restrict__ d, int n) {
  for (int i = blockIdx.x * blockDim.x + threadIdx.x; i < n;
       i += gridDim.x * blockDim.x)
    d[i] = f2bf(s[i]);
}

// ---------------- LayerNorm (optionally fused shift+window gather) -------
// one wave32 per token; each lane owns 8 channels of C=256
__global__ __launch_bounds__(256)
void ln_kernel(const float* __restrict__ xin, const float* __restrict__ scale,
               const float* __restrict__ bias, unsigned short* __restrict__ outb,
               int shifted) {
  int wv = threadIdx.x >> 5, lane = threadIdx.x & 31;
  int r = blockIdx.x * 8 + wv;
  unsigned srcRow = shifted ? swin_pix(r) : (unsigned)r;
  const float* p = xin + (size_t)srcRow * CH + lane * 8;
  float v[8];
  *(float4*)&v[0] = *(const float4*)(p);
  *(float4*)&v[4] = *(const float4*)(p + 4);
  float s = 0.f;
#pragma unroll
  for (int e = 0; e < 8; e++) s += v[e];
#pragma unroll
  for (int m = 16; m >= 1; m >>= 1) s += __shfl_xor(s, m, 32);
  float mu = s * (1.f / 256.f);
  float vs = 0.f;
#pragma unroll
  for (int e = 0; e < 8; e++) { float d = v[e] - mu; vs += d * d; }
#pragma unroll
  for (int m = 16; m >= 1; m >>= 1) vs += __shfl_xor(vs, m, 32);
  float inv = rsqrtf(vs * (1.f / 256.f) + 1e-5f);
  int cbase = lane * 8;
  unsigned short o[8];
#pragma unroll
  for (int e = 0; e < 8; e++)
    o[e] = f2bf((v[e] - mu) * inv * scale[cbase + e] + bias[cbase + e]);
  *(u32x4*)(outb + (size_t)r * CH + cbase) = *(u32x4*)o;
}

// ---------------- generic bf16 WMMA GEMM: out = epi(A[M,K] @ W[N,K]^T + b)
// K loop manually unrolled x2 with two STATICALLY-NAMED fragment sets so all
// array indices are compile-time constants (SROA keeps them in VGPRs; a
// runtime-indexed double buffer spilled to scratch).  Requires K % 64 == 0.
// epi 0: store bf16          epi 1: GELU(exact), store bf16
// epi 2: +resid, store f32   epi 3: +resid with swin un-shift remap, f32
#define LOAD_SLICE(AF, BF, kk)                                 \
  do {                                                         \
    AF.u[0] = *(const u32x4*)(a0 + (kk));                      \
    AF.u[1] = *(const u32x4*)(a1 + (kk));                      \
    _Pragma("unroll") for (int nt = 0; nt < 4; nt++) {         \
      BF[nt].u[0] = *(const u32x4*)(wb[nt] + (kk));            \
      BF[nt].u[1] = *(const u32x4*)(wb[nt] + (kk) + 8);        \
    }                                                          \
  } while (0)

#define WMMA_SLICE(AF, BF)                                     \
  do {                                                         \
    _Pragma("unroll") for (int nt = 0; nt < 4; nt++)           \
        acc[nt] = __builtin_amdgcn_wmma_f32_16x16x32_bf16(     \
            false, AF.v, false, BF[nt].v, (short)0, acc[nt],   \
            false, false);                                     \
  } while (0)

__global__ __launch_bounds__(256)
void gemm_kernel(const unsigned short* __restrict__ A,
                 const unsigned short* __restrict__ W,
                 const float* __restrict__ bias, void* __restrict__ outp,
                 const float* __restrict__ resid, int M, int N, int K, int epi) {
  int wv = threadIdx.x >> 5, lane = threadIdx.x & 31;
  int half = lane >> 4, l16 = lane & 15;
  int mBase = blockIdx.x * 128 + wv * 16;
  int nBase = blockIdx.y * 64;

  const unsigned short* a0 = A + (size_t)(mBase + l16) * K + half * 16;  // u[0]
  const unsigned short* a1 = a0 + 8;                                     // u[1]
  const unsigned short* wb[4];
#pragma unroll
  for (int nt = 0; nt < 4; nt++)
    wb[nt] = W + (size_t)(nBase + nt * 16 + l16) * K + half * 16;

  Frag aE, aO;
  Frag bE[4], bO[4];
  LOAD_SLICE(aE, bE, 0);

  f32x8 acc[4] = {};
  for (int k = 0; k < K; k += 64) {
    LOAD_SLICE(aO, bO, k + 32);                  // in flight during E WMMAs
    __builtin_prefetch(a0 + k + 64, 0, 1);       // global_prefetch_b8
    WMMA_SLICE(aE, bE);
    if (k + 64 < K)
      LOAD_SLICE(aE, bE, k + 64);                // in flight during O WMMAs
    WMMA_SLICE(aO, bO);
  }

#pragma unroll
  for (int nt = 0; nt < 4; nt++) {
#pragma unroll
    for (int e = 0; e < 8; e++) {
      int row = mBase + e + half * 8;            // C/D layout (ISA 7.12.2)
      int col = nBase + nt * 16 + l16;
      float vl = acc[nt][e] + bias[col];
      if (epi == 0) {
        ((unsigned short*)outp)[(size_t)row * N + col] = f2bf(vl);
      } else if (epi == 1) {
        float g = 0.5f * vl * (1.f + erff(vl * 0.70710678118654752f));
        ((unsigned short*)outp)[(size_t)row * N + col] = f2bf(g);
      } else if (epi == 2) {
        size_t idx = (size_t)row * N + col;
        ((float*)outp)[idx] = resid[idx] + vl;
      } else {
        size_t idx = (size_t)swin_pix(row) * CH + col;
        ((float*)outp)[idx] = resid[idx] + vl;
      }
    }
  }
}

// ---------------- windowed attention: 1 block / window, 1 wave / head ----
__global__ __launch_bounds__(256)
void attn_kernel(const unsigned short* __restrict__ qkv,
                 const float* __restrict__ relb,
                 unsigned short* __restrict__ outb) {
  __shared__ unsigned short VT[CH * VT_STRIDE];      // V^T [ch][token], 36KB
  __shared__ unsigned short PS[8 * 16 * PS_STRIDE];  // per-wave P strip, 18KB

  int win = blockIdx.x, tid = threadIdx.x;
  // stage V^T: thread owns channel `tid` (global reads coalesce across lanes)
  {
    const unsigned short* vsrc = qkv + (size_t)win * 64 * QKVN + 512 + tid;
#pragma unroll 8
    for (int t = 0; t < 64; t++) VT[tid * VT_STRIDE + t] = vsrc[(size_t)t * QKVN];
  }
  __syncthreads();

  int head = tid >> 5, lane = tid & 31, half = lane >> 4, l16 = lane & 15;
  int wi = win & 63, wh = wi >> 3, ww = wi & 7;
  const float scale = 0.17677669529663687f;    // 1/sqrt(32)
  unsigned pbase = head * 16 * PS_STRIDE;

  // K-matrix B-fragments: identical for every mi strip -> load once per head
  Frag bk[4];
#pragma unroll
  for (int ni = 0; ni < 4; ni++) {
    const unsigned short* krow =
        qkv + (size_t)(win * 64 + ni * 16 + l16) * QKVN + 256 + head * 32 +
        half * 16;
    bk[ni].u[0] = *(const u32x4*)(krow);
    bk[ni].u[1] = *(const u32x4*)(krow + 8);
  }
  // V B-fragments (from padded LDS): also mi-invariant -> hoist
  Frag bv[2][2];
#pragma unroll
  for (int nt = 0; nt < 2; nt++)
#pragma unroll
    for (int kt = 0; kt < 2; kt++) {
      const unsigned short* vr =
          &VT[(head * 32 + nt * 16 + l16) * VT_STRIDE + kt * 32 + half * 16];
      bv[nt][kt].u[0] = *(const u32x4*)(vr);
      bv[nt][kt].u[1] = *(const u32x4*)(vr + 8);
    }

  for (int mi = 0; mi < 4; mi++) {
    // ---- S = Q K^T  (M=16 strip, N=64, K=32) ----
    const unsigned short* qrow =
        qkv + (size_t)(win * 64 + mi * 16 + l16) * QKVN + head * 32;
    Frag a;
    a.u[0] = *(const u32x4*)(qrow + half * 16);
    a.u[1] = *(const u32x4*)(qrow + half * 16 + 8);
    f32x8 s[4] = {};
#pragma unroll
    for (int ni = 0; ni < 4; ni++)
      s[ni] = __builtin_amdgcn_wmma_f32_16x16x32_bf16(
          false, a.v, false, bk[ni].v, (short)0, s[ni], false, false);

    // ---- scale + relative-position bias + shift mask ----
#pragma unroll
    for (int e = 0; e < 8; e++) {
      int n = mi * 16 + e + half * 8;
      int i1 = n >> 3, j1 = n & 7;
      int y1 = wh * 8 + i1, x1 = ww * 8 + j1;
      int c1 = ((y1 < 56) ? 0 : ((y1 < 60) ? 1 : 2)) * 3 +
               ((x1 < 56) ? 0 : ((x1 < 60) ? 1 : 2));
#pragma unroll
      for (int ni = 0; ni < 4; ni++) {
        int m = ni * 16 + l16;
        int i2 = m >> 3, j2 = m & 7;
        int y2 = wh * 8 + i2, x2 = ww * 8 + j2;
        int c2 = ((y2 < 56) ? 0 : ((y2 < 60) ? 1 : 2)) * 3 +
                 ((x2 < 56) ? 0 : ((x2 < 60) ? 1 : 2));
        int ridx = (i1 - i2 + 7) * 15 + (j1 - j2 + 7);
        float bse = relb[ridx * 8 + head];
        s[ni][e] = s[ni][e] * scale + bse + ((c1 != c2) ? -100.f : 0.f);
      }
    }
    // ---- row softmax (row spans 4 tiles x 16 lanes of this half) ----
#pragma unroll
    for (int e = 0; e < 8; e++) {
      float mx = fmaxf(fmaxf(s[0][e], s[1][e]), fmaxf(s[2][e], s[3][e]));
#pragma unroll
      for (int m = 8; m >= 1; m >>= 1) mx = fmaxf(mx, __shfl_xor(mx, m, 32));
      float sm = 0.f;
#pragma unroll
      for (int ni = 0; ni < 4; ni++) {
        float ev = __expf(s[ni][e] - mx);
        s[ni][e] = ev;
        sm += ev;
      }
#pragma unroll
      for (int m = 8; m >= 1; m >>= 1) sm += __shfl_xor(sm, m, 32);
      float inv = 1.f / sm;
#pragma unroll
      for (int ni = 0; ni < 4; ni++) s[ni][e] *= inv;
    }
    // ---- repack P (C-layout regs -> A-layout) through LDS ----
    __syncthreads();   // prior iteration's PS reads are done block-wide
#pragma unroll
    for (int e = 0; e < 8; e++) {
      int rr = e + half * 8;
#pragma unroll
      for (int ni = 0; ni < 4; ni++)
        PS[pbase + rr * PS_STRIDE + ni * 16 + l16] = f2bf(s[ni][e]);
    }
    __syncthreads();
    // ---- O = P V  (M=16, N=32, K=64 in two 32-steps) ----
    Frag ap[2];
#pragma unroll
    for (int kt = 0; kt < 2; kt++) {
      const unsigned short* pr =
          &PS[pbase + l16 * PS_STRIDE + kt * 32 + half * 16];
      ap[kt].u[0] = *(const u32x4*)(pr);
      ap[kt].u[1] = *(const u32x4*)(pr + 8);
    }
#pragma unroll
    for (int nt = 0; nt < 2; nt++) {
      f32x8 d = {};
#pragma unroll
      for (int kt = 0; kt < 2; kt++)
        d = __builtin_amdgcn_wmma_f32_16x16x32_bf16(
            false, ap[kt].v, false, bv[nt][kt].v, (short)0, d, false, false);
#pragma unroll
      for (int e = 0; e < 8; e++) {
        int tok = win * 64 + mi * 16 + e + half * 8;
        int c = head * 32 + nt * 16 + l16;
        outb[(size_t)tok * CH + c] = f2bf(d[e]);
      }
    }
  }
}

// ---------------- launcher ----------------
extern "C" void kernel_launch(void* const* d_in, const int* in_sizes, int n_in,
                              void* d_out, int out_size, void* d_ws,
                              size_t ws_size, hipStream_t stream) {
  const float* x      = (const float*)d_in[0];
  const float* ln1_s  = (const float*)d_in[1];
  const float* ln1_b  = (const float*)d_in[2];
  const float* qkv_w  = (const float*)d_in[3];
  const float* qkv_b  = (const float*)d_in[4];
  const float* proj_w = (const float*)d_in[5];
  const float* proj_b = (const float*)d_in[6];
  const float* rel_b  = (const float*)d_in[7];
  const float* ln2_s  = (const float*)d_in[8];
  const float* ln2_b  = (const float*)d_in[9];
  const float* w1     = (const float*)d_in[10];
  const float* b1     = (const float*)d_in[11];
  const float* w2     = (const float*)d_in[12];
  const float* b2     = (const float*)d_in[13];

  char* ws = (char*)d_ws;
  unsigned short* wq   = (unsigned short*)(ws);                 // 768x256 bf16
  unsigned short* wp   = (unsigned short*)(ws + 393216);        // 256x256
  unsigned short* w1b  = (unsigned short*)(ws + 524288);        // 1024x256
  unsigned short* w2b  = (unsigned short*)(ws + 1048576);       // 256x1024
  unsigned short* hbuf = (unsigned short*)(ws + 1572864);       // 65536x256 bf16
  unsigned short* qkvb = (unsigned short*)(ws + 35127296);      // 65536x768 bf16 (region 128MB, reused as `mid`)
  unsigned short* attn = (unsigned short*)(ws + 169345024);     // 65536x256 bf16
  float*          x2   = (float*)(ws + 202899456);              // 65536x256 f32
  unsigned short* mid  = qkvb;                                  // reuse
  unsigned short* h2   = hbuf;                                  // reuse
  float* out = (float*)d_out;

  // weights -> bf16
  cvt_kernel<<<192, 256, 0, stream>>>(qkv_w, wq, QKVN * CH);
  cvt_kernel<<<64, 256, 0, stream>>>(proj_w, wp, CH * CH);
  cvt_kernel<<<256, 256, 0, stream>>>(w1, w1b, HIDN * CH);
  cvt_kernel<<<256, 256, 0, stream>>>(w2, w2b, CH * HIDN);

  // LN1 + roll(-4,-4) + window partition
  ln_kernel<<<TOK / 8, 256, 0, stream>>>(x, ln1_s, ln1_b, hbuf, 1);
  // QKV
  gemm_kernel<<<dim3(TOK / 128, QKVN / 64), 256, 0, stream>>>(
      hbuf, wq, qkv_b, qkvb, nullptr, TOK, QKVN, CH, 0);
  // windowed attention
  attn_kernel<<<NWIN, 256, 0, stream>>>(qkvb, rel_b, attn);
  // proj + un-shift + residual -> x2 (f32)
  gemm_kernel<<<dim3(TOK / 128, CH / 64), 256, 0, stream>>>(
      attn, wp, proj_b, x2, x, TOK, CH, CH, 3);
  // LN2
  ln_kernel<<<TOK / 8, 256, 0, stream>>>(x2, ln2_s, ln2_b, h2, 0);
  // MLP
  gemm_kernel<<<dim3(TOK / 128, HIDN / 64), 256, 0, stream>>>(
      h2, w1b, b1, mid, nullptr, TOK, HIDN, CH, 1);
  gemm_kernel<<<dim3(TOK / 128, CH / 64), 256, 0, stream>>>(
      mid, w2b, b2, out, x2, TOK, CH, HIDN, 2);
}
